// DisambiguateCandidates_28132035789009
// MI455X (gfx1250) — matrix operand order repaired and checked by reading, not compiled, not run on hardware
//
#include <hip/hip_runtime.h>

typedef __attribute__((ext_vector_type(16))) _Float16 v16h;
typedef __attribute__((ext_vector_type(8)))  float    v8f;

#define NPTS 131072
#define NB   32
#define NCAND 4
#define SCORE_STRIDE 5   // cols 0..3 = candidate scores, col 4 = point count
#define PPT 4            // points per thread (chunks of K=32 per wave)

// ---------------------------------------------------------------------------
// small helpers: 3x3 determinant and compile-time cofactor of a 4x4
// ---------------------------------------------------------------------------
__device__ __forceinline__ float det3(float a00,float a01,float a02,
                                      float a10,float a11,float a12,
                                      float a20,float a21,float a22){
  return a00*(a11*a22 - a12*a21)
       - a01*(a10*a22 - a12*a20)
       + a02*(a10*a21 - a11*a20);
}

template<int R, int C>
__device__ __forceinline__ float cofT(const float m[4][4]){
  constexpr int r0 = (R==0)?1:0;
  constexpr int r1 = (R<=1)?2:1;
  constexpr int r2 = (R<=2)?3:2;
  constexpr int c0 = (C==0)?1:0;
  constexpr int c1 = (C<=1)?2:1;
  constexpr int c2 = (C<=2)?3:2;
  float d = det3(m[r0][c0],m[r0][c1],m[r0][c2],
                 m[r1][c0],m[r1][c1],m[r1][c2],
                 m[r2][c0],m[r2][c1],m[r2][c2]);
  return (((R+C)&1) ? -d : d);
}

// ---------------------------------------------------------------------------
// kernel 0: zero the score table (32 batches x 5 columns of f32)
// ---------------------------------------------------------------------------
__global__ void DisambiguateCandidates_zero_ws(float* ws){
  ws[threadIdx.x] = 0.0f;   // launched with exactly NB*SCORE_STRIDE threads
}

// ---------------------------------------------------------------------------
// per-point cheirality mask (bit c = candidate c passes both depth tests)
// ---------------------------------------------------------------------------
__device__ __forceinline__ unsigned cheirality_mask(
    const float* __restrict__ Ropt, const float* __restrict__ topt,
    int b, float x1n, float y1n, float x2n, float y2n)
{
  unsigned fmask = 0u;
  for (int c = 0; c < NCAND; ++c) {
    const float* R = Ropt + (b*NCAND + c)*9;
    const float* t = topt + (b*NCAND + c)*3;
    const float R00=R[0],R01=R[1],R02=R[2];
    const float R10=R[3],R11=R[4],R12=R[5];
    const float R20=R[6],R21=R[7],R22=R[8];
    const float t0=t[0], t1=t[1], t2=t[2];

    // DLT rows 2,3:  a2 = x2n*P2row2 - P2row0 ; a3 = y2n*P2row2 - P2row1
    const float a20 = x2n*R20 - R00, a21 = x2n*R21 - R01,
                a22 = x2n*R22 - R02, a23 = x2n*t2  - t0;
    const float a30 = y2n*R20 - R10, a31 = y2n*R21 - R11,
                a32 = y2n*R22 - R12, a33 = y2n*t2  - t1;

    // M = A^T A  (rows 0,1 are sparse: (-1,0,x1n,0) and (0,-1,y1n,0))
    float m[4][4];
    m[0][0] = 1.0f + a20*a20 + a30*a30;
    m[0][1] =        a20*a21 + a30*a31;
    m[0][2] = -x1n + a20*a22 + a30*a32;
    m[0][3] =        a20*a23 + a30*a33;
    m[1][1] = 1.0f + a21*a21 + a31*a31;
    m[1][2] = -y1n + a21*a22 + a31*a32;
    m[1][3] =        a21*a23 + a31*a33;
    m[2][2] = x1n*x1n + y1n*y1n + a22*a22 + a32*a32;
    m[2][3] =        a22*a23 + a32*a33;
    m[3][3] =        a23*a23 + a33*a33;
    m[1][0]=m[0][1]; m[2][0]=m[0][2]; m[3][0]=m[0][3];
    m[2][1]=m[1][2]; m[3][1]=m[1][3]; m[3][2]=m[2][3];

    // null direction of near-rank-3 M: best column of adj(M)
    const float d0 = cofT<0,0>(m), d1 = cofT<1,1>(m),
                d2 = cofT<2,2>(m), d3 = cofT<3,3>(m);
    int   js = 0; float dm = fabsf(d0);
    if (fabsf(d1) > dm) { dm = fabsf(d1); js = 1; }
    if (fabsf(d2) > dm) { dm = fabsf(d2); js = 2; }
    if (fabsf(d3) > dm) {                 js = 3; }

    float v0,v1,v2,v3;
    switch (js) {
      case 0: v0=cofT<0,0>(m); v1=cofT<0,1>(m); v2=cofT<0,2>(m); v3=cofT<0,3>(m); break;
      case 1: v0=cofT<1,0>(m); v1=cofT<1,1>(m); v2=cofT<1,2>(m); v3=cofT<1,3>(m); break;
      case 2: v0=cofT<2,0>(m); v1=cofT<2,1>(m); v2=cofT<2,2>(m); v3=cofT<2,3>(m); break;
      default:v0=cofT<3,0>(m); v1=cofT<3,1>(m); v2=cofT<3,2>(m); v3=cofT<3,3>(m); break;
    }

    // sign-only cheirality: z1>0 <=> v2*v3>0 ; z2>0 <=> (P2row2.v)*v3>0
    const float z2dot = R20*v0 + R21*v1 + R22*v2 + t2*v3;
    const bool in_front = (v2*v3 > 0.0f) && (z2dot*v3 > 0.0f);
    fmask |= (in_front ? 1u : 0u) << c;
  }
  return fmask;
}

// ---------------------------------------------------------------------------
// kernel 1: per-point scoring + WMMA one-hot segmented reduction.
//   grid = NPTS/(256*PPT) blocks x 256 threads; each wave32 owns PPT
//   consecutive chunks of 32 sorted points and accumulates the one-hot
//   matmul S += Ind(16x32) x F(32x16) in f32 C tiles before one scatter.
// ---------------------------------------------------------------------------
__global__ void __launch_bounds__(256)
DisambiguateCandidates_score(const float* __restrict__ Ropt,   // [32][4][3][3]
                             const float* __restrict__ topt,   // [32][4][3]
                             const float* __restrict__ Kb,     // [32][3][3]
                             const float* __restrict__ p1,     // [N][2]
                             const float* __restrict__ p2,     // [N][2]
                             const int*   __restrict__ bidx,   // [N]
                             float* __restrict__ scores)       // ws: [32][5]
{
  const int lane  = threadIdx.x & 31;
  const int wgid  = (blockIdx.x * 256 + threadIdx.x) >> 5;  // global wave id
  const int base  = wgid * (32 * PPT);                      // first point of wave

  const int half = lane >> 4;      // which 16-lane half of the wave
  const int rlow = lane & 15;
  const int kbase = (lane < 16) ? 0 : 16;

  // K-slot index map for the f16 A-operand layout (constant per lane)
  int ksl[16];
#pragma unroll
  for (int e = 0; e < 16; ++e) {
    const int v = e >> 1, lo = e & 1;
    ksl[e] = ((v < 4) ? 2*v : 2*(v-4) + 16) + half*8 + lo;
  }

  v8f c0 = {}, c1 = {};            // accumulators: batches 0..15 / 16..31

  for (int it = 0; it < PPT; ++it) {
    const int pt = base + it*32 + lane;

    // uniform prefetch of next chunk's point data (global_prefetch_b8)
    if (it + 1 < PPT) {
      __builtin_prefetch(&p1[2*(pt + 32)], 0, 1);
      __builtin_prefetch(&p2[2*(pt + 32)], 0, 1);
      __builtin_prefetch(&bidx[pt + 32],   0, 1);
    }

    // ---- per-point geometry ----------------------------------------------
    const int   b  = bidx[pt];
    const float x1 = p1[2*pt+0], y1 = p1[2*pt+1];
    const float x2 = p2[2*pt+0], y2 = p2[2*pt+1];

    const float fx = Kb[b*9+0], cx = Kb[b*9+2];
    const float fy = Kb[b*9+4], cy = Kb[b*9+5];
    const float x1n = (x1-cx)/fx, y1n = (y1-cy)/fy;
    const float x2n = (x2-cx)/fx, y2n = (y2-cy)/fy;

    const unsigned fmask = cheirality_mask(Ropt, topt, b, x1n, y1n, x2n, y2n);

    // ---- build f16 operands via cross-lane shuffles ----------------------
    v16h a0, a1, bop;
#pragma unroll
    for (int e = 0; e < 16; ++e) {
      const int bk = __shfl(b, ksl[e], 32);
      a0[e] = (bk == rlow     ) ? (_Float16)1.0f : (_Float16)0.0f;
      a1[e] = (bk == rlow + 16) ? (_Float16)1.0f : (_Float16)0.0f;
    }
#pragma unroll
    for (int e = 0; e < 16; ++e) {
      const unsigned fm = (unsigned)__shfl((int)fmask, (kbase >> 1) + ((e < 8) ? e : e + 8), 32);
      float val = 0.0f;
      if (rlow < 4)       val = ((fm >> rlow) & 1u) ? 1.0f : 0.0f;
      else if (rlow == 4) val = 1.0f;   // counts column
      bop[e] = (_Float16)val;
    }

    // S += Ind x F  (exact: 0/1 f16 inputs, integer f32 accumulation)
    c0 = __builtin_amdgcn_wmma_f32_16x16x32_f16(false, a0, false, bop,
                                                (short)0, c0, false, false);
    c1 = __builtin_amdgcn_wmma_f32_16x16x32_f16(false, a1, false, bop,
                                                (short)0, c1, false, false);
  }

  // ---- one sparse scatter per wave (sorted bidx => 1-2 batches live) -----
  if (rlow < SCORE_STRIDE) {
    const int mbase = (lane < 16) ? 0 : 8;
#pragma unroll
    for (int r = 0; r < 8; ++r) {
      const float s0 = c0[r];
      if (s0 != 0.0f) atomicAdd(&scores[(mbase + r)      *SCORE_STRIDE + rlow], s0);
      const float s1 = c1[r];
      if (s1 != 0.0f) atomicAdd(&scores[(16 + mbase + r) *SCORE_STRIDE + rlow], s1);
    }
  }
}

// ---------------------------------------------------------------------------
// kernel 2: per-batch argmax + gather best R/t/candidate
//   out layout: [0,288) best_R, [288,384) best_t, [384,416) best_candidate
// ---------------------------------------------------------------------------
__global__ void DisambiguateCandidates_finalize(const float* __restrict__ scores,
                                                const float* __restrict__ Ropt,
                                                const float* __restrict__ topt,
                                                float* __restrict__ out)
{
  const int b = threadIdx.x;
  if (b >= NB) return;
  float bs  = scores[b*SCORE_STRIDE + 0];
  int best  = 0;
#pragma unroll
  for (int c = 1; c < NCAND; ++c) {
    const float s = scores[b*SCORE_STRIDE + c];
    if (s > bs) { bs = s; best = c; }          // first-max (argmax semantics)
  }
  const bool has = scores[b*SCORE_STRIDE + 4] > 0.0f;

#pragma unroll
  for (int i = 0; i < 9; ++i)
    out[b*9 + i] = has ? Ropt[(b*NCAND + best)*9 + i] : 0.0f;
#pragma unroll
  for (int i = 0; i < 3; ++i)
    out[NB*9 + b*3 + i] = has ? topt[(b*NCAND + best)*3 + i] : 0.0f;

  ((int*)out)[NB*9 + NB*3 + b] = has ? best : 0;
}

// ---------------------------------------------------------------------------
extern "C" void kernel_launch(void* const* d_in, const int* in_sizes, int n_in,
                              void* d_out, int out_size, void* d_ws, size_t ws_size,
                              hipStream_t stream)
{
  const float* Ropt = (const float*)d_in[0];  // 32*4*3*3
  const float* topt = (const float*)d_in[1];  // 32*4*3
  const float* Kb   = (const float*)d_in[2];  // 32*3*3
  const float* p1   = (const float*)d_in[3];  // N*2
  const float* p2   = (const float*)d_in[4];  // N*2
  const int*   bi   = (const int*)  d_in[5];  // N
  float* ws = (float*)d_ws;                   // 32*5 f32 score/count table

  DisambiguateCandidates_zero_ws<<<1, NB*SCORE_STRIDE, 0, stream>>>(ws);
  DisambiguateCandidates_score<<<NPTS/(256*PPT), 256, 0, stream>>>(Ropt, topt,
                                                                   Kb, p1, p2,
                                                                   bi, ws);
  DisambiguateCandidates_finalize<<<1, 32, 0, stream>>>(ws, Ropt, topt,
                                                        (float*)d_out);
}
